// GatedAttentionUnit_23184233463842
// MI455X (gfx1250) — matrix operand order repaired and checked by reading, not compiled, not run on hardware
//
#include <hip/hip_runtime.h>
#include <hip/hip_bf16.h>
#include <math.h>

typedef __attribute__((ext_vector_type(16))) _Float16 v16h;
typedef __attribute__((ext_vector_type(8)))  float    v8f;
typedef __attribute__((ext_vector_type(8)))  _Float16 h8;
typedef __attribute__((ext_vector_type(4)))  _Float16 h4;

#define BATCH_ 2
#define S_     4096
#define H_     768
#define E_     1536
#define D_     128
#define NI_    (2*E_ + D_)   /* 3200 */
#define MROWS_ 8192          /* BATCH_*S_ */

// ---------------- elementwise f32 -> f16 (vectorized) ----------------
__global__ void cvt_f32_f16(const float4* __restrict__ in, h4* __restrict__ out, long n4) {
  long i = (long)blockIdx.x * blockDim.x + threadIdx.x;
  long stride = (long)gridDim.x * blockDim.x;
  for (; i < n4; i += stride) {
    float4 v = in[i];
    h4 o; o.x = (_Float16)v.x; o.y = (_Float16)v.y; o.z = (_Float16)v.z; o.w = (_Float16)v.w;
    out[i] = o;
  }
}

// ---------------- fused convert + transpose: fp32 RxC -> f16 CxR ----------------
__global__ __launch_bounds__(256) void cvt_transpose(const float* __restrict__ in,
                                                     _Float16* __restrict__ out,
                                                     int R, int C) {
  __shared__ float tile[32][33];
  const int ct = blockIdx.x * 32, rt = blockIdx.y * 32;
  const int tx = threadIdx.x & 31, ty = threadIdx.x >> 5;   // 32 x 8
  #pragma unroll
  for (int yy = 0; yy < 32; yy += 8)
    tile[ty + yy][tx] = in[(size_t)(rt + ty + yy) * C + ct + tx];
  __syncthreads();
  #pragma unroll
  for (int yy = 0; yy < 32; yy += 8)
    out[(size_t)(ct + ty + yy) * R + rt + tx] = (_Float16)tile[tx][ty + yy];
}

// ---------------- RoPE: q/k from qk, fold attention scale into q ----------------
__global__ void rope_kernel(const float* __restrict__ qk,
                            const float* __restrict__ sinb, const float* __restrict__ cosb,
                            const float* __restrict__ qs,   const float* __restrict__ ks,
                            _Float16* __restrict__ q, _Float16* __restrict__ k,
                            float attnScale) {
  int row = blockIdx.x;          // 0 .. B*S-1
  int s   = row & (S_ - 1);
  int i   = threadIdx.x;         // 0..63 pair index
  float x1 = qk[(size_t)row * D_ + 2*i];
  float x2 = qk[(size_t)row * D_ + 2*i + 1];
  float sv = sinb[(size_t)s * (D_/2) + i];
  float cv = cosb[(size_t)s * (D_/2) + i];
  float q1 = x1 * qs[2*i], q2 = x2 * qs[2*i+1];
  float k1 = x1 * ks[2*i], k2 = x2 * ks[2*i+1];
  q[(size_t)row * D_ + i]          = (_Float16)((q1*cv - q2*sv) * attnScale);
  q[(size_t)row * D_ + D_/2 + i]   = (_Float16)((q1*sv + q2*cv) * attnScale);
  k[(size_t)row * D_ + i]          = (_Float16)(k1*cv - k2*sv);
  k[(size_t)row * D_ + D_/2 + i]   = (_Float16)(k1*sv + k2*cv);
}

// ---------------- row softmax: fp32 scores -> f16 probabilities ----------------
__global__ __launch_bounds__(256) void softmax_rows(const float* __restrict__ scores,
                                                    _Float16* __restrict__ A) {
  __shared__ float red[256];
  const int row = blockIdx.x;
  const int t = threadIdx.x;
  const float4* r4 = (const float4*)(scores + (size_t)row * S_);   // 1024 float4 per row
  float4 v[4];
  float m = -1e30f;
  #pragma unroll
  for (int j = 0; j < 4; j++) {
    v[j] = r4[t + j*256];
    m = fmaxf(m, fmaxf(fmaxf(v[j].x, v[j].y), fmaxf(v[j].z, v[j].w)));
  }
  red[t] = m; __syncthreads();
  for (int o = 128; o > 0; o >>= 1) { if (t < o) red[t] = fmaxf(red[t], red[t+o]); __syncthreads(); }
  m = red[0]; __syncthreads();
  float4 e[4];
  float sum = 0.f;
  #pragma unroll
  for (int j = 0; j < 4; j++) {
    e[j].x = __expf(v[j].x - m); e[j].y = __expf(v[j].y - m);
    e[j].z = __expf(v[j].z - m); e[j].w = __expf(v[j].w - m);
    sum += (e[j].x + e[j].y) + (e[j].z + e[j].w);
  }
  red[t] = sum; __syncthreads();
  for (int o = 128; o > 0; o >>= 1) { if (t < o) red[t] += red[t+o]; __syncthreads(); }
  float inv = __builtin_amdgcn_rcpf(red[0]);
  h4* outp = (h4*)(A + (size_t)row * S_);
  #pragma unroll
  for (int j = 0; j < 4; j++) {
    h4 o;
    o.x = (_Float16)(e[j].x * inv); o.y = (_Float16)(e[j].y * inv);
    o.z = (_Float16)(e[j].z * inv); o.w = (_Float16)(e[j].w * inv);
    outp[t + j*256] = o;
  }
}

// ---------------- 128x128 WMMA GEMM, double-buffered LDS, B always NxK ----------------
// C = A(MxK row-major) @ Bt^T, where Bt is NxK row-major (leading dim ldb).
// EPI 0: fp32 store to outF.
// EPI 1: silu; block-uniform split: u/v written TRANSPOSED (E x MROWS_) as h8, qk fp32 row-major.
// EPI 2: gate-multiply with uT (h8 loads), f16 row-major store.
struct FragH { union { v16h v; float4 f4[2]; }; };

#define LSTR 40   // LDS row stride in halfs (8-half pad: bank spread + 16B alignment)

template<int EPI>
__global__ __launch_bounds__(256)
void gemm_wmma(const _Float16* __restrict__ Abase, const _Float16* __restrict__ Btbase,
               int N, int K, int lda, int ldb,
               long sA, long sB, long sO,
               float* __restrict__ outF, _Float16* __restrict__ outH,
               _Float16* __restrict__ uT, _Float16* __restrict__ vT, float* __restrict__ qkOut,
               const _Float16* __restrict__ gateT)
{
  __shared__ __align__(16) _Float16 lA[2][128 * LSTR];
  __shared__ __align__(16) _Float16 lB[2][128 * LSTR];

  const int z = blockIdx.z;
  const _Float16* Ag = Abase  + (size_t)z * sA;
  const _Float16* Bg = Btbase + (size_t)z * sB;

  const int blockM = blockIdx.y * 128;
  const int blockN = blockIdx.x * 128;

  const int t    = threadIdx.x;
  const int lane = t & 31;
  const int wid  = t >> 5;
  const int waveM = wid >> 2;   // 0..1  -> 64 rows each
  const int waveN = wid & 3;    // 0..3  -> 32 cols each

  const int rT = t >> 1, cT = (t & 1) * 16;   // tile coords: 128 rows x 32 halfs

  v8f acc[4][2];
  #pragma unroll
  for (int i = 0; i < 4; i++)
    #pragma unroll
    for (int j = 0; j < 2; j++)
      #pragma unroll
      for (int e = 0; e < 8; e++) acc[i][j][e] = 0.f;

  // ---- prologue: tile kt=0 into buffer 0 ----
  {
    const _Float16* gp = Ag + (size_t)(blockM + rT) * lda + cT;
    *(float4*)&lA[0][rT*LSTR + cT]     = *(const float4*)gp;
    *(float4*)&lA[0][rT*LSTR + cT + 8] = *(const float4*)(gp + 8);
    const _Float16* bp = Bg + (size_t)(blockN + rT) * ldb + cT;
    *(float4*)&lB[0][rT*LSTR + cT]     = *(const float4*)bp;
    *(float4*)&lB[0][rT*LSTR + cT + 8] = *(const float4*)(bp + 8);
  }
  __syncthreads();

  int cur = 0;
  for (int kt = 0; kt < K; kt += 32) {
    // ---- prefetch next tile into registers (VMEM overlaps WMMA below) ----
    const bool pre = (kt + 32) < K;
    float4 pa0{}, pa1{}, pb0{}, pb1{};
    if (pre) {
      const _Float16* gp = Ag + (size_t)(blockM + rT) * lda + (kt + 32) + cT;
      pa0 = *(const float4*)gp;
      pa1 = *(const float4*)(gp + 8);
      const _Float16* bp = Bg + (size_t)(blockN + rT) * ldb + (kt + 32) + cT;
      pb0 = *(const float4*)bp;
      pb1 = *(const float4*)(bp + 8);
    }

    // ---- fragments per ISA 16-bit wave32 layouts + 8 WMMAs ----
    {
      FragH aF[4], bF[2];
      const int h = lane >> 4, l16 = lane & 15;
      #pragma unroll
      for (int i = 0; i < 4; i++) {
        int m = waveM*64 + i*16 + l16;
        aF[i].f4[0] = *(const float4*)&lA[cur][m*LSTR + h*8];        // K = h*8 .. +7
        aF[i].f4[1] = *(const float4*)&lA[cur][m*LSTR + 16 + h*8];   // K = 16 + h*8 .. +7
      }
      #pragma unroll
      for (int j = 0; j < 2; j++) {
        int n = waveN*32 + j*16 + l16;
        bF[j].f4[0] = *(const float4*)&lB[cur][n*LSTR + h*8];        // K = h*8 .. +7
        bF[j].f4[1] = *(const float4*)&lB[cur][n*LSTR + 16 + h*8];   // K = 16 + h*8 .. +7
      }
      #pragma unroll
      for (int i = 0; i < 4; i++)
        #pragma unroll
        for (int j = 0; j < 2; j++)
          acc[i][j] = __builtin_amdgcn_wmma_f32_16x16x32_f16(
              false, aF[i].v, false, bF[j].v, (short)0, acc[i][j], false, false);
    }

    // ---- commit prefetched tile into the other buffer ----
    if (pre) {
      *(float4*)&lA[cur^1][rT*LSTR + cT]     = pa0;
      *(float4*)&lA[cur^1][rT*LSTR + cT + 8] = pa1;
      *(float4*)&lB[cur^1][rT*LSTR + cT]     = pb0;
      *(float4*)&lB[cur^1][rT*LSTR + cT + 8] = pb1;
    }
    __syncthreads();
    cur ^= 1;
  }

  // ---- epilogue: C/D layout -> lane l16 = N col, VGPR r -> M row (r + 8*h) ----
  const int h = lane >> 4, l16 = lane & 15;
  #pragma unroll
  for (int i = 0; i < 4; i++) {
    #pragma unroll
    for (int j = 0; j < 2; j++) {
      const int gn = blockN + waveN*32 + j*16 + l16;
      const int mb = blockM + waveM*64 + i*16 + 8*h;   // multiple of 8
      if (EPI == 0) {
        #pragma unroll
        for (int r = 0; r < 8; r++)
          outF[(size_t)z*sO + (size_t)(mb + r)*N + gn] = acc[i][j][r];
      } else if (EPI == 1) {
        // block-uniform region select (boundaries 1536/3072 are multiples of 128)
        if (blockN < 2*E_) {
          h8 o;
          #pragma unroll
          for (int r = 0; r < 8; r++) {
            float x = acc[i][j][r];
            o[r] = (_Float16)(x * __builtin_amdgcn_rcpf(1.f + __expf(-x)));
          }
          if (blockN < E_)   *(h8*)&uT[(size_t)gn        * MROWS_ + mb] = o;   // u^T: E x 8192
          else               *(h8*)&vT[(size_t)(gn - E_) * MROWS_ + mb] = o;   // v^T: E x 8192
        } else {
          #pragma unroll
          for (int r = 0; r < 8; r++) {
            float x = acc[i][j][r];
            qkOut[(size_t)(mb + r)*D_ + (gn - 2*E_)] = x * __builtin_amdgcn_rcpf(1.f + __expf(-x));
          }
        }
      } else {   // EPI == 2: o = u * (A@v); gateT is u^T (E x 8192), rows here are per-batch
        h8 g = *(const h8*)&gateT[(size_t)gn * MROWS_ + (size_t)z * S_ + mb];
        #pragma unroll
        for (int r = 0; r < 8; r++)
          outH[(size_t)z*sO + (size_t)(mb + r)*N + gn] = (_Float16)((float)g[r] * acc[i][j][r]);
      }
    }
  }
}

extern "C" void kernel_launch(void* const* d_in, const int* in_sizes, int n_in,
                              void* d_out, int out_size, void* d_ws, size_t ws_size,
                              hipStream_t stream) {
  (void)in_sizes; (void)n_in; (void)out_size; (void)ws_size;
  const float* hidden = (const float*)d_in[0];
  const float* sinb   = (const float*)d_in[1];
  const float* cosb   = (const float*)d_in[2];
  const float* W_i    = (const float*)d_in[3];
  const float* W_o    = (const float*)d_in[4];
  const float* qs     = (const float*)d_in[5];
  const float* ks     = (const float*)d_in[6];
  float* out = (float*)d_out;

  const long M = MROWS_;   // 8192 rows
  char* p = (char*)d_ws;
  auto carve = [&](size_t bytes) -> void* {
    void* r = (void*)p; p += (bytes + 255) & ~(size_t)255; return r;
  };
  _Float16* hid16 = (_Float16*)carve((size_t)M * H_ * 2);
  _Float16* WiT   = (_Float16*)carve((size_t)H_ * NI_ * 2);   // NI_ x H_
  _Float16* WoT   = (_Float16*)carve((size_t)E_ * H_ * 2);    // H_ x E_
  _Float16* uT    = (_Float16*)carve((size_t)M * E_ * 2);     // E_ x M
  _Float16* vT    = (_Float16*)carve((size_t)M * E_ * 2);     // E_ x M
  float*    qk32  = (float*)   carve((size_t)M * D_ * 4);
  _Float16* q16   = (_Float16*)carve((size_t)M * D_ * 2);
  _Float16* k16   = (_Float16*)carve((size_t)M * D_ * 2);
  float*    sc32  = (float*)   carve((size_t)BATCH_ * S_ * S_ * 4);
  _Float16* A16   = (_Float16*)carve((size_t)BATCH_ * S_ * S_ * 2);
  _Float16* o16   = (_Float16*)carve((size_t)M * E_ * 2);

  cvt_f32_f16<<<1024, 256, 0, stream>>>((const float4*)hidden, (h4*)hid16, M * H_ / 4);
  cvt_transpose<<<dim3(NI_/32, H_/32), 256, 0, stream>>>(W_i, WiT, H_, NI_);   // -> NI_ x H_
  cvt_transpose<<<dim3(H_/32, E_/32),  256, 0, stream>>>(W_o, WoT, E_, H_);    // -> H_ x E_

  // x = silu(hidden @ W_i) -> uT, vT (f16, transposed), qk (f32 row-major)
  gemm_wmma<1><<<dim3(NI_/128, M/128, 1), 256, 0, stream>>>(
      hid16, WiT, NI_, H_, H_, H_, 0, 0, 0,
      nullptr, nullptr, uT, vT, qk32, nullptr);

  const float attnScale = (logf(4096.f) / logf(512.f)) / sqrtf(128.f);
  rope_kernel<<<(int)M, 64, 0, stream>>>(qk32, sinb, cosb, qs, ks, q16, k16, attnScale);

  // scores = q @ k^T (k is N x K layout), per batch
  gemm_wmma<0><<<dim3(S_/128, S_/128, BATCH_), 256, 0, stream>>>(
      q16, k16, S_, D_, D_, D_, (long)S_*D_, (long)S_*D_, (long)S_*S_,
      sc32, nullptr, nullptr, nullptr, nullptr, nullptr);

  softmax_rows<<<(int)M, 256, 0, stream>>>(sc32, A16);

  // o = u * (A @ v): B = vT (E_ x 8192), batch selects column window z*S_
  gemm_wmma<2><<<dim3(E_/128, S_/128, BATCH_), 256, 0, stream>>>(
      A16, vT + 0, E_, S_, S_, MROWS_, (long)S_*S_, (long)S_, (long)S_*E_,
      nullptr, o16, nullptr, nullptr, nullptr, uT);

  // out = o @ W_o (fp32 result): B = WoT (H_ x E_)
  gemm_wmma<0><<<dim3(H_/128, M/128, 1), 256, 0, stream>>>(
      o16, WoT, H_, E_, E_, E_, 0, 0, 0,
      out, nullptr, nullptr, nullptr, nullptr, nullptr);
}
